// NaiveAttention_8392366096962
// MI455X (gfx1250) — matrix operand order repaired and checked by reading, not compiled, not run on hardware
//
#include <hip/hip_runtime.h>
#include <hip/hip_bf16.h>
#include <stdint.h>

// ---------------------------------------------------------------------------
// NaiveAttention for MI455X (gfx1250): bf16 WMMA GEMMs + flash-attention core.
// Double-buffered LDS staging, async TDM (tensor_load_to_lds) for V tiles,
// exp2-domain online softmax with scale folded into the Q projection.
//   H=2048, NH=16, HD=128, B=2, T=2048.  fp32 in/out, bf16 compute, f32 accum.
// ---------------------------------------------------------------------------

typedef __attribute__((ext_vector_type(16))) __bf16 v16bf;
typedef __attribute__((ext_vector_type(8)))  __bf16 bf16x8;
typedef __attribute__((ext_vector_type(4)))  __bf16 bf16x4;
typedef __attribute__((ext_vector_type(8)))  float  v8f;
typedef __attribute__((ext_vector_type(4)))  float  v4f;
typedef __attribute__((ext_vector_type(4)))  unsigned int u32x4;
typedef __attribute__((ext_vector_type(4)))  int    i32x4;
typedef __attribute__((ext_vector_type(8)))  int    i32x8;

static constexpr int   kH  = 2048;
static constexpr int   kNH = 16;
static constexpr int   kHD = 128;
static constexpr int   kB  = 2;
static constexpr int   kT  = 2048;
static constexpr int   kBT = kB * kT;  // 4096 rows
// softmax scale folded into Q projection, pre-multiplied by log2(e) so the
// online softmax runs in the exp2 domain (v_exp_f32 is base-2 natively).
static constexpr float kQScale = 0.08838834764831845f * 1.4426950408889634f;

#if defined(__has_builtin)
#if __has_builtin(__builtin_amdgcn_tensor_load_to_lds) && \
    __has_builtin(__builtin_amdgcn_s_wait_tensorcnt)
#define USE_TDM 1
#endif
#if __has_builtin(__builtin_amdgcn_exp2f)
#define HAVE_EXP2 1
#endif
#endif

static __device__ __forceinline__ float fexp2(float x) {
#ifdef HAVE_EXP2
  return __builtin_amdgcn_exp2f(x);
#else
  return exp2f(x);
#endif
}

static __device__ __forceinline__ v8f wmma_bf16(v16bf a, v16bf b, v8f c) {
  // D = A(16x32 bf16) * B(32x16 bf16) + C(16x16 f32)
  return __builtin_amdgcn_wmma_f32_16x16x32_bf16(false, a, false, b,
                                                 (short)0, c, false, false);
}

static __device__ __forceinline__ v16bf join16(bf16x8 lo, bf16x8 hi) {
  return __builtin_shufflevector(lo, hi, 0, 1, 2, 3, 4, 5, 6, 7,
                                 8, 9, 10, 11, 12, 13, 14, 15);
}

// ---------------------------------------------------------------------------
// fp32 -> bf16 conversion (4 elements / thread)
// ---------------------------------------------------------------------------
__global__ void cvt_f32_to_bf16(const float* __restrict__ src,
                                __bf16* __restrict__ dst, int n) {
  int i = (blockIdx.x * blockDim.x + threadIdx.x) * 4;
  if (i < n) {
    v4f v = *reinterpret_cast<const v4f*>(src + i);
    bf16x4 o;
#pragma unroll
    for (int e = 0; e < 4; ++e) o[e] = (__bf16)v[e];
    *reinterpret_cast<bf16x4*>(dst + i) = o;
  }
}

// ---------------------------------------------------------------------------
// C[M,N] = ((A[M,K] @ W[N,K]^T) + bias[N]) * outScale.  A,W bf16 row-major.
// Block: 128 thr / 4 waves -> 64x64 tile; W k-slab double-buffered in LDS.
// ---------------------------------------------------------------------------
template <typename OutT>
__global__ __launch_bounds__(128) void gemm_nt_bias(
    const __bf16* __restrict__ A, const __bf16* __restrict__ W,
    const float* __restrict__ bias, OutT* __restrict__ C,
    int M, int N, int K, float outScale) {
  __shared__ __align__(16) __bf16 Wt[2][32][80];  // [buf][k][n], 160B rows

  const int tid  = threadIdx.x;
  const int lane = tid & 31;
  const int wave = tid >> 5;
  const int m0   = blockIdx.x * 64 + wave * 16;
  const int n0   = blockIdx.y * 64;
  const int mrow = lane & 15;
  const int half = lane >> 4;
  const int ncol = lane & 15;

  v8f acc[4];
#pragma unroll
  for (int t = 0; t < 4; ++t) acc[t] = (v8f)0.0f;

  const __bf16* arow = A + (size_t)(m0 + mrow) * K;
  // staging role: thread covers one n-row (64 rows, 2 thr/row) x 16 k values
  const int srow  = tid >> 1;
  const int kbase = (tid & 1) * 16;
  const __bf16* wrow = W + (size_t)(n0 + srow) * K + kbase;

  auto stageW = [&](int k0, int buf) {
    bf16x8 w0 = *reinterpret_cast<const bf16x8*>(wrow + k0);
    bf16x8 w1 = *reinterpret_cast<const bf16x8*>(wrow + k0 + 8);
#pragma unroll
    for (int e = 0; e < 8; ++e) {
      Wt[buf][kbase + e][srow]     = w0[e];
      Wt[buf][kbase + 8 + e][srow] = w1[e];
    }
  };

  stageW(0, 0);
  __syncthreads();

  int buf = 0;
  for (int k0 = 0; k0 < K; k0 += 32, buf ^= 1) {
    if (k0 + 32 < K) stageW(k0 + 32, buf ^ 1);  // overlap with compute

    v16bf Av = join16(
        *reinterpret_cast<const bf16x8*>(arow + k0 + 8 * half),
        *reinterpret_cast<const bf16x8*>(arow + k0 + 16 + 8 * half));
#pragma unroll
    for (int nt = 0; nt < 4; ++nt) {
      v16bf Bv;
#pragma unroll
      for (int e = 0; e < 16; ++e)
        Bv[e] = Wt[buf][e + 8 * half + (e & 8)][16 * nt + ncol];
      acc[nt] = wmma_bf16(Av, Bv, acc[nt]);
    }
    __syncthreads();
  }

#pragma unroll
  for (int nt = 0; nt < 4; ++nt) {
#pragma unroll
    for (int v = 0; v < 8; ++v) {
      const int m = m0 + v + 8 * half;   // C layout: M = vgpr + 8*(lane/16)
      const int n = n0 + 16 * nt + ncol; // N = lane%16
      C[(size_t)m * N + n] = (OutT)((acc[nt][v] + bias[n]) * outScale);
    }
  }
}

// ---------------------------------------------------------------------------
// Online-softmax update for a 16x32 score tile (exp2 domain).
// Rows live in 16-lane halves of each C VGPR; MASK specializes the causal path.
// ---------------------------------------------------------------------------
template <bool MASK>
static __device__ __forceinline__ void softmax_update(
    v8f S0, v8f S1, int key0, int q0, int half, int ncol,
    float* m_i, float* l_i, float* alpha, __bf16* pw /* Plds[wave] */) {
#pragma unroll
  for (int v = 0; v < 8; ++v) {
    const int qg = q0 + v + 8 * half;
    float s0 = S0[v];
    float s1 = S1[v];
    if (MASK) {
      if (key0 + ncol > qg)      s0 = -__builtin_inff();
      if (key0 + 16 + ncol > qg) s1 = -__builtin_inff();
    }
    float rmax = fmaxf(s0, s1);
#pragma unroll
    for (int off = 8; off > 0; off >>= 1)
      rmax = fmaxf(rmax, __shfl_xor(rmax, off, 32));
    const float mn = fmaxf(m_i[v], rmax);  // finite after first block
    const float p0 = fexp2(s0 - mn);
    const float p1 = fexp2(s1 - mn);
    float rsum = p0 + p1;
#pragma unroll
    for (int off = 8; off > 0; off >>= 1)
      rsum += __shfl_xor(rsum, off, 32);
    alpha[v] = fexp2(m_i[v] - mn);
    l_i[v]   = l_i[v] * alpha[v] + rsum;
    m_i[v]   = mn;
    __bf16* prow = pw + (v + 8 * half) * 40;
    prow[ncol]      = (__bf16)p0;
    prow[16 + ncol] = (__bf16)p1;
  }
}

// ---------------------------------------------------------------------------
// Flash attention (causal). Block = 128 thr / 4 waves = 64 query rows of one
// (batch, head). K/V double-buffered in LDS; V staged by async TDM.
// ---------------------------------------------------------------------------
__global__ __launch_bounds__(128) void flash_attn_causal(
    const __bf16* __restrict__ Q, const __bf16* __restrict__ K,
    const __bf16* __restrict__ V, __bf16* __restrict__ ctx) {
  __shared__ __align__(16) __bf16 Kt[2][128][40];    // [buf][d][key]   20 KB
  __shared__ __align__(16) __bf16 Vs[2][32][136];    // [buf][key][d]   17 KB
  __shared__ __align__(16) __bf16 Plds[4][16 * 40];  // per-wave P tile  5 KB

  const int tid  = threadIdx.x;
  const int lane = tid & 31;
  const int wave = tid >> 5;
  const int qt   = blockIdx.x & 31;            // 64-row query tile (T/64 = 32)
  const int h    = (blockIdx.x >> 5) & (kNH - 1);
  const int b    = blockIdx.x >> 9;
  const int q0   = qt * 64 + wave * 16;
  const int mrow = lane & 15;
  const int half = lane >> 4;
  const int ncol = lane & 15;

  const size_t headoff = ((size_t)b * kT) * kH + (size_t)h * kHD;

  // Q as WMMA A-tiles (already scaled by 1/sqrt(HD)*log2e at projection)
  v16bf Aq[4];
  {
    const __bf16* qrow = Q + headoff + (size_t)(q0 + mrow) * kH;
#pragma unroll
    for (int s = 0; s < 4; ++s)
      Aq[s] = join16(
          *reinterpret_cast<const bf16x8*>(qrow + 32 * s + 8 * half),
          *reinterpret_cast<const bf16x8*>(qrow + 32 * s + 16 + 8 * half));
  }

  v8f acc[8];
  float m_i[8], l_i[8];
#pragma unroll
  for (int t = 0; t < 8; ++t) {
    acc[t] = (v8f)0.0f;
    m_i[t] = -__builtin_inff();
    l_i[t] = 0.0f;
  }

  // staging roles
  const int skey = tid >> 2;         // key row 0..31
  const int sdb  = (tid & 3) * 32;   // 32-wide d slice
  const __bf16* kgrow = K + headoff + (size_t)skey * kH + sdb;

  auto stageK = [&](int key0, int buf) {
    const __bf16* kg = kgrow + (size_t)key0 * kH;
#pragma unroll
    for (int c = 0; c < 4; ++c) {
      bf16x8 kv = *reinterpret_cast<const bf16x8*>(kg + 8 * c);
#pragma unroll
      for (int e = 0; e < 8; ++e) Kt[buf][sdb + 8 * c + e][skey] = kv[e];
    }
  };

#ifdef USE_TDM
  // TDM descriptor skeleton: V tile 128x32 bf16, row stride H, LDS pad of
  // 4 dwords per 64-dword row -> Vs[.][32][136] layout. (ISA 08_async §8)
  i32x8 g1;
  {
    const unsigned dim0 = 128, dim1 = 32, tile0 = 128, tile1 = 32;
    g1[0] = (int)((1u << 16)          // data_size = 2 bytes
                  | (1u << 20)        // pad_enable
                  | (5u << 22)        // pad_interval: every 64 dwords
                  | (3u << 25));      // pad_amount: 4 dwords
    g1[1] = (int)(dim0 << 16);        // tensor_dim0[15:0] @ bits 63:48
    g1[2] = (int)((dim0 >> 16) | (dim1 << 16));
    g1[3] = (int)((dim1 >> 16) | (tile0 << 16));
    g1[4] = (int)(tile1);
    g1[5] = (int)(unsigned)kH;        // tensor_dim0_stride[31:0]
    g1[6] = 0;
    g1[7] = 0;
  }
  const i32x4 gz4 = (i32x4)0;
  const i32x8 gz8 = (i32x8)0;
  const unsigned vlds[2] = {(unsigned)(uintptr_t)(&Vs[0][0][0]),
                            (unsigned)(uintptr_t)(&Vs[1][0][0])};
#endif

  auto issueV = [&](int key0, int buf) {
#ifdef USE_TDM
    if (wave == 0) {
      const uint64_t ga = (uint64_t)(uintptr_t)(V + headoff + (size_t)key0 * kH);
      u32x4 g0;
      g0[0] = 1u;                                       // count=1
      g0[1] = vlds[buf];                                // lds_addr
      g0[2] = (unsigned)(ga & 0xFFFFFFFFu);             // global_addr lo
      g0[3] = (unsigned)((ga >> 32) & 0x01FFFFFFu) | (2u << 30);  // hi|type=2
      __builtin_amdgcn_tensor_load_to_lds(g0, g1, gz4, gz4, gz8, 0);
    }
#else
    const __bf16* vg = V + headoff + (size_t)(key0 + skey) * kH + sdb;
#pragma unroll
    for (int c = 0; c < 4; ++c)
      *reinterpret_cast<bf16x8*>(&Vs[buf][skey][sdb + 8 * c]) =
          *reinterpret_cast<const bf16x8*>(vg + 8 * c);
#endif
  };

  const int nkb = 2 * qt + 2;  // 32-key blocks covering rows q0_blk..+63

  // prologue: stage block 0 into buffer 0
  stageK(0, 0);
  issueV(0, 0);
#ifdef USE_TDM
  if (wave == 0) __builtin_amdgcn_s_wait_tensorcnt((short)0);
#endif
  __syncthreads();

  int buf = 0;
  for (int j = 0; j < nkb; ++j, buf ^= 1) {
    const int key0 = j * 32;

    // ---- stage block j+1 into the other buffer (overlaps compute) ----
    if (j + 1 < nkb) {
      stageK(key0 + 32, buf ^ 1);
      issueV(key0 + 32, buf ^ 1);
      if (j + 2 < nkb)
        __builtin_prefetch(kgrow + (size_t)(key0 + 64) * kH, 0, 1);
    }

    // ---- compute block j (skip if fully masked for this wave's rows) ----
    if (key0 <= q0 + 15) {
      v8f S0 = (v8f)0.0f, S1 = (v8f)0.0f;
#pragma unroll
      for (int s = 0; s < 4; ++s) {
        v16bf Bk0, Bk1;
#pragma unroll
        for (int e = 0; e < 16; ++e) {
          const int d = 32 * s + e + 8 * half + (e & 8);
          Bk0[e] = Kt[buf][d][ncol];
          Bk1[e] = Kt[buf][d][16 + ncol];
        }
        S0 = wmma_bf16(Aq[s], Bk0, S0);
        S1 = wmma_bf16(Aq[s], Bk1, S1);
      }

      float alpha[8];
      if (key0 + 31 > q0)  // diagonal block: causal masking needed
        softmax_update<true>(S0, S1, key0, q0, half, ncol, m_i, l_i, alpha,
                             &Plds[wave][0]);
      else                 // interior block: mask-free fast path
        softmax_update<false>(S0, S1, key0, q0, half, ncol, m_i, l_i, alpha,
                              &Plds[wave][0]);

#pragma unroll
      for (int t = 0; t < 8; ++t) {
#pragma unroll
        for (int v = 0; v < 8; ++v) acc[t][v] = acc[t][v] * alpha[v];
      }

      // wave-local DS drain before cross-lane P reload (DS in-order per wave)
      asm volatile("s_wait_dscnt 0x0" ::: "memory");

      v16bf Ap;
      {
        const __bf16* prow = &Plds[wave][mrow * 40];
        Ap = join16(*reinterpret_cast<const bf16x8*>(prow + 8 * half),
                    *reinterpret_cast<const bf16x8*>(prow + 16 + 8 * half));
      }
#pragma unroll
      for (int nt = 0; nt < 8; ++nt) {
        v16bf Bv;
#pragma unroll
        for (int e = 0; e < 16; ++e)
          Bv[e] = Vs[buf][e + 8 * half + (e & 8)][16 * nt + ncol];
        acc[nt] = wmma_bf16(Ap, Bv, acc[nt]);
      }
    }

#ifdef USE_TDM
    if (wave == 0) __builtin_amdgcn_s_wait_tensorcnt((short)0);  // V(j+1) done
#endif
    __syncthreads();  // K(j+1) stores + all buffer reads complete
  }

  // ---- normalize and write context (bf16) ----
  float inv[8];
#pragma unroll
  for (int v = 0; v < 8; ++v) inv[v] = 1.0f / l_i[v];
#pragma unroll
  for (int nt = 0; nt < 8; ++nt) {
#pragma unroll
    for (int v = 0; v < 8; ++v) {
      ctx[headoff + (size_t)(q0 + v + 8 * half) * kH + 16 * nt + ncol] =
          (__bf16)(acc[nt][v] * inv[v]);
    }
  }
}

// ---------------------------------------------------------------------------
extern "C" void kernel_launch(void* const* d_in, const int* in_sizes, int n_in,
                              void* d_out, int out_size, void* d_ws, size_t ws_size,
                              hipStream_t stream) {
  (void)in_sizes; (void)n_in; (void)out_size; (void)ws_size;
  const float* x  = (const float*)d_in[0];
  const float* wq = (const float*)d_in[1];
  const float* bq = (const float*)d_in[2];
  const float* wk = (const float*)d_in[3];
  const float* bk = (const float*)d_in[4];
  const float* wv = (const float*)d_in[5];
  const float* bv = (const float*)d_in[6];
  const float* wo = (const float*)d_in[7];
  const float* bo = (const float*)d_in[8];

  char* ws = (char*)d_ws;
  size_t off = 0;
  auto alloc = [&](size_t bytes) -> void* {
    void* p = ws + off;
    off += (bytes + 255) & ~(size_t)255;
    return p;
  };
  const size_t actBytes = (size_t)kBT * kH * sizeof(__bf16);  // 16 MiB
  const size_t wBytes   = (size_t)kH * kH * sizeof(__bf16);   // 8 MiB
  __bf16* xb  = (__bf16*)alloc(actBytes);
  __bf16* wqb = (__bf16*)alloc(wBytes);
  __bf16* wkb = (__bf16*)alloc(wBytes);
  __bf16* wvb = (__bf16*)alloc(wBytes);
  __bf16* wob = (__bf16*)alloc(wBytes);
  __bf16* Qb  = (__bf16*)alloc(actBytes);
  __bf16* Kb  = (__bf16*)alloc(actBytes);
  __bf16* Vb  = (__bf16*)alloc(actBytes);
  __bf16* Cb  = (__bf16*)alloc(actBytes);

  const int tc = 256;
  const int nAct = kBT * kH;
  const int nW   = kH * kH;
  cvt_f32_to_bf16<<<(nAct / 4 + tc - 1) / tc, tc, 0, stream>>>(x,  xb,  nAct);
  cvt_f32_to_bf16<<<(nW / 4 + tc - 1) / tc, tc, 0, stream>>>(wq, wqb, nW);
  cvt_f32_to_bf16<<<(nW / 4 + tc - 1) / tc, tc, 0, stream>>>(wk, wkb, nW);
  cvt_f32_to_bf16<<<(nW / 4 + tc - 1) / tc, tc, 0, stream>>>(wv, wvb, nW);
  cvt_f32_to_bf16<<<(nW / 4 + tc - 1) / tc, tc, 0, stream>>>(wo, wob, nW);

  dim3 gg(kBT / 64, kH / 64);  // 64 x 32 blocks of 128 threads
  // Q carries the softmax scale (exp2 domain); K/V/out are unscaled.
  gemm_nt_bias<__bf16><<<gg, 128, 0, stream>>>(xb, wqb, bq, Qb, kBT, kH, kH,
                                               kQScale);
  gemm_nt_bias<__bf16><<<gg, 128, 0, stream>>>(xb, wkb, bk, Kb, kBT, kH, kH,
                                               1.0f);
  gemm_nt_bias<__bf16><<<gg, 128, 0, stream>>>(xb, wvb, bv, Vb, kBT, kH, kH,
                                               1.0f);

  const int nBlk = kB * kNH * (kT / 64);  // 1024 blocks
  flash_attn_causal<<<nBlk, 128, 0, stream>>>(Qb, Kb, Vb, Cb);

  gemm_nt_bias<float><<<gg, 128, 0, stream>>>(Cb, wob, bo, (float*)d_out,
                                              kBT, kH, kH, 1.0f);
}
